// KNNConsistencyLoss_25615184953631
// MI455X (gfx1250) — matrix operand order repaired and checked by reading, not compiled, not run on hardware
//
#include <hip/hip_runtime.h>

#define B_N   4096
#define D_S   64
#define D_T   768
#define KNN   10
#define INVT  10.0f   // 1/TEMP, TEMP=0.1

typedef __attribute__((ext_vector_type(16))) __bf16 v16bf;
typedef __attribute__((ext_vector_type(8)))  __bf16 v8bf;
typedef __attribute__((ext_vector_type(8)))  float  v8f;
typedef __attribute__((ext_vector_type(2)))  float  v2f;

// ---------------------------------------------------------------- utilities
__global__ void zero_acc_kernel(float* acc) {
    if (threadIdx.x < 8) acc[threadIdx.x] = 0.0f;
}

// Normalize teacher rows, store as bf16 (feeds bf16 WMMA).
__global__ void normalize_teacher_kernel(const float* __restrict__ te,
                                         unsigned short* __restrict__ tn_bits) {
    __shared__ float red[256];
    int row = blockIdx.x;
    float s = 0.0f;
    for (int c = threadIdx.x; c < D_T; c += 256) {
        float v = te[(size_t)row * D_T + c];
        s += v * v;
    }
    red[threadIdx.x] = s;
    __syncthreads();
    for (int o = 128; o > 0; o >>= 1) {
        if (threadIdx.x < o) red[threadIdx.x] += red[threadIdx.x + o];
        __syncthreads();
    }
    float inv = 1.0f / fmaxf(sqrtf(red[0]), 1e-8f);
    __bf16* tn = (__bf16*)tn_bits;
    for (int c = threadIdx.x; c < D_T; c += 256) {
        tn[(size_t)row * D_T + c] = (__bf16)(te[(size_t)row * D_T + c] * inv);
    }
}

// Student A-operand with negated time column (Lorentz metric folded into A).
__global__ void prep_student_a_kernel(const float* __restrict__ st, float* __restrict__ sa) {
    int i = blockIdx.x * blockDim.x + threadIdx.x;
    if (i < B_N * D_S) {
        float v = st[i];
        sa[i] = ((i & (D_S - 1)) == 0) ? -v : v;
    }
}

// ------------------------------------------------ bf16 fragment load helpers
// A 16x32 bf16 fragment for row-strip base `rowp` (stride D_T), K-offset k0.
// Lane layout: halves 0..7 -> K = k0 + 8*hi + (0..7); halves 8..15 -> +16.
__device__ __forceinline__ v16bf load_afrag_bf16(const __bf16* rowp, int k0, int hi) {
    v8bf c0 = *(const v8bf*)(rowp + k0 + hi * 8);
    v8bf c1 = *(const v8bf*)(rowp + k0 + 16 + hi * 8);
    v16bf a;
#pragma unroll
    for (int i = 0; i < 8; i++) { a[i] = c0[i]; a[i + 8] = c1[i]; }
    return a;
}
// B 32x16 bf16 fragment: halves 0..7 -> K = k0 + 8*hi + (0..7); halves 8..15 -> +16.
__device__ __forceinline__ v16bf load_bfrag_bf16(const __bf16* rowp, int k0, int hi) {
    v8bf c0 = *(const v8bf*)(rowp + k0 + hi * 8);
    v8bf c1 = *(const v8bf*)(rowp + k0 + 16 + hi * 8);
    v16bf b;
#pragma unroll
    for (int i = 0; i < 8; i++) { b[i] = c0[i]; b[i + 8] = c1[i]; }
    return b;
}

__device__ __forceinline__ v8f wmma_bf16(v16bf a, v16bf b, v8f c) {
    return __builtin_amdgcn_wmma_f32_16x16x32_bf16(false, a, false, b,
                                                   (short)0, c, false, false);
}

// ---------------------------------------------------- teacher Gram (bf16 WMMA)
// 2x2 register-blocked: one wave computes a 32x32 output tile -> 4 WMMAs per
// K-step from 4 fragment loads (1 fragment per WMMA instead of 2).
__global__ void teacher_gram_kernel(const unsigned short* __restrict__ tn_bits,
                                    float* __restrict__ Dt) {
    const __bf16* tn = (const __bf16*)tn_bits;
    int wave = (blockIdx.x * blockDim.x + threadIdx.x) >> 5;
    int lane = threadIdx.x & 31;
    int wm = wave >> 7;          // 128 wave-tiles per dim (32 rows each)
    int wn = wave & 127;
    int hi = lane >> 4;
    int lr = lane & 15;

    const __bf16* a0p = tn + (size_t)(wm * 32 + lr) * D_T;
    const __bf16* a1p = tn + (size_t)(wm * 32 + 16 + lr) * D_T;
    const __bf16* b0p = tn + (size_t)(wn * 32 + lr) * D_T;
    const __bf16* b1p = tn + (size_t)(wn * 32 + 16 + lr) * D_T;

    v8f acc00 = {}, acc01 = {}, acc10 = {}, acc11 = {};
    for (int k0 = 0; k0 < D_T; k0 += 32) {
        v16bf a0 = load_afrag_bf16(a0p, k0, hi);
        v16bf a1 = load_afrag_bf16(a1p, k0, hi);
        v16bf b0 = load_bfrag_bf16(b0p, k0, hi);
        v16bf b1 = load_bfrag_bf16(b1p, k0, hi);
        acc00 = wmma_bf16(a0, b0, acc00);
        acc01 = wmma_bf16(a0, b1, acc01);
        acc10 = wmma_bf16(a1, b0, acc10);
        acc11 = wmma_bf16(a1, b1, acc11);
    }

    int row0 = wm * 32, col0 = wn * 32;
#pragma unroll
    for (int r = 0; r < 8; r++) {
        int rr = r + hi * 8;
        Dt[(size_t)(row0 + rr) * B_N + (col0 + lr)]           = 1.0f - acc00[r];
        Dt[(size_t)(row0 + rr) * B_N + (col0 + 16 + lr)]      = 1.0f - acc01[r];
        Dt[(size_t)(row0 + 16 + rr) * B_N + (col0 + lr)]      = 1.0f - acc10[r];
        Dt[(size_t)(row0 + 16 + rr) * B_N + (col0 + 16 + lr)] = 1.0f - acc11[r];
    }
}

// ------------------------------------------------ student Gram (fp32 WMMA)
// Same 2x2 blocking; K=64 in steps of 4 via v_wmma_f32_16x16x4_f32.
// D_s = arccosh(max(-inner, 1+1e-7)).
__device__ __forceinline__ v8f wmma_f32(v2f a, v2f b, v8f c) {
    return __builtin_amdgcn_wmma_f32_16x16x4_f32(false, a, false, b,
                                                 (short)0, c, false, false);
}

__global__ void student_gram_kernel(const float* __restrict__ sa,
                                    const float* __restrict__ sb,
                                    float* __restrict__ Ds) {
    int wave = (blockIdx.x * blockDim.x + threadIdx.x) >> 5;
    int lane = threadIdx.x & 31;
    int wm = wave >> 7;
    int wn = wave & 127;
    int hi = lane >> 4;
    int lr = lane & 15;

    const float* a0p = sa + (size_t)(wm * 32 + lr) * D_S;
    const float* a1p = sa + (size_t)(wm * 32 + 16 + lr) * D_S;
    const float* b0p = sb + (size_t)(wn * 32 + lr) * D_S;
    const float* b1p = sb + (size_t)(wn * 32 + 16 + lr) * D_S;

    v8f acc00 = {}, acc01 = {}, acc10 = {}, acc11 = {};
    for (int k0 = 0; k0 < D_S; k0 += 4) {
        v2f a0 = *(const v2f*)(a0p + k0 + hi * 2);
        v2f a1 = *(const v2f*)(a1p + k0 + hi * 2);
        v2f b0 = *(const v2f*)(b0p + k0 + hi * 2);
        v2f b1 = *(const v2f*)(b1p + k0 + hi * 2);
        acc00 = wmma_f32(a0, b0, acc00);
        acc01 = wmma_f32(a0, b1, acc01);
        acc10 = wmma_f32(a1, b0, acc10);
        acc11 = wmma_f32(a1, b1, acc11);
    }

    int row0 = wm * 32, col0 = wn * 32;
#pragma unroll
    for (int r = 0; r < 8; r++) {
        int rr = r + hi * 8;
        Ds[(size_t)(row0 + rr) * B_N + (col0 + lr)] =
            acoshf(fmaxf(-acc00[r], 1.0f + 1e-7f));
        Ds[(size_t)(row0 + rr) * B_N + (col0 + 16 + lr)] =
            acoshf(fmaxf(-acc01[r], 1.0f + 1e-7f));
        Ds[(size_t)(row0 + 16 + rr) * B_N + (col0 + lr)] =
            acoshf(fmaxf(-acc10[r], 1.0f + 1e-7f));
        Ds[(size_t)(row0 + 16 + rr) * B_N + (col0 + 16 + lr)] =
            acoshf(fmaxf(-acc11[r], 1.0f + 1e-7f));
    }
}

// ------------------------------------- per-row top-10 (self excluded) + KL loss
__global__ void topk_kl_kernel(const float* __restrict__ Dt,
                               const float* __restrict__ Ds,
                               float* __restrict__ acc) {
    const float FINF = 3.0e38f;
    int r = blockIdx.x;
    int lane = threadIdx.x & 31;

    float cand[KNN];
    int   cidx[KNN];
#pragma unroll
    for (int i = 0; i < KNN; i++) { cand[i] = FINF; cidx[i] = 0; }

    const float* row = Dt + (size_t)r * B_N;
    for (int j = lane; j < B_N; j += 32) {
        if (j == r) continue;                 // drop self (reference drops nearest = diag)
        float v = row[j];
        if (v < cand[KNN - 1]) {
            int p = KNN - 1;
            while (p > 0 && cand[p - 1] > v) {
                cand[p] = cand[p - 1]; cidx[p] = cidx[p - 1]; p--;
            }
            cand[p] = v; cidx[p] = j;
        }
    }

    // merge 32 sorted lists: 10 rounds of wave-min
    float selD[KNN];
    int   selI[KNN];
    int p = 0;
    for (int t = 0; t < KNN; t++) {
        float h = (p < KNN) ? cand[p] : FINF;
        float m = h;
        for (int o = 16; o > 0; o >>= 1) m = fminf(m, __shfl_xor(m, o, 32));
        unsigned long long bal = __ballot(h == m);
        int src = __ffsll((unsigned long long)bal) - 1;
        int mine = cidx[(p < KNN) ? p : (KNN - 1)];
        int bi = __shfl(mine, src, 32);
        if (lane == src) p++;
        selD[t] = m;
        selI[t] = bi;
    }

    if (lane == 0) {
        float lt[KNN], ls[KNN];
        float mt = -FINF, ms = -FINF;
        const float* srow = Ds + (size_t)r * B_N;
#pragma unroll
        for (int t = 0; t < KNN; t++) {
            lt[t] = -selD[t] * INVT;
            ls[t] = -srow[selI[t]] * INVT;
            mt = fmaxf(mt, lt[t]);
            ms = fmaxf(ms, ls[t]);
        }
        float st = 0.0f, ss = 0.0f;
#pragma unroll
        for (int t = 0; t < KNN; t++) {
            st += __expf(lt[t] - mt);
            ss += __expf(ls[t] - ms);
        }
        float lse_t = mt + __logf(st);
        float lse_s = ms + __logf(ss);
        float kl = 0.0f;
#pragma unroll
        for (int t = 0; t < KNN; t++) {
            float logPt = lt[t] - lse_t;
            float Pt = __expf(logPt);
            kl += Pt * (logPt - (ls[t] - lse_s));
        }
        atomicAdd(&acc[0], kl);
    }
}

// ---------------------------------- global pair loss: single-pass 5 moments
__global__ void global_pairs_kernel(const int* __restrict__ i1,
                                    const int* __restrict__ i2,
                                    const float* __restrict__ Dt,
                                    const float* __restrict__ Ds,
                                    float* __restrict__ acc, int N) {
    float s0 = 0, s1 = 0, s2 = 0, s3 = 0, s4 = 0;
    for (int i = blockIdx.x * blockDim.x + threadIdx.x; i < N;
         i += gridDim.x * blockDim.x) {
        int a = i1[i], b = i2[i];
        size_t o = (size_t)a * B_N + b;
        float ds = Ds[o], dt = Dt[o];
        s0 += ds; s1 += dt; s2 += ds * ds; s3 += dt * dt; s4 += ds * dt;
    }
    for (int o = 16; o > 0; o >>= 1) {
        s0 += __shfl_xor(s0, o, 32);
        s1 += __shfl_xor(s1, o, 32);
        s2 += __shfl_xor(s2, o, 32);
        s3 += __shfl_xor(s3, o, 32);
        s4 += __shfl_xor(s4, o, 32);
    }
    if ((threadIdx.x & 31) == 0) {
        atomicAdd(&acc[1], s0);
        atomicAdd(&acc[2], s1);
        atomicAdd(&acc[3], s2);
        atomicAdd(&acc[4], s3);
        atomicAdd(&acc[5], s4);
    }
}

__global__ void finalize_kernel(const float* __restrict__ acc, int NG,
                                float* __restrict__ out) {
    float local = acc[0] / (float)B_N;
    float n = (float)NG;
    float mds = acc[1] / n, mdt = acc[2] / n;
    float scale = (mds + 1e-8f) / (mdt + 1e-8f);
    float glob = (acc[3] - 2.0f * scale * acc[5] + scale * scale * acc[4]) / n;
    float tot = local + 0.5f * glob;
    if (!isfinite(tot)) tot = 0.0f;
    out[0] = tot;
}

// --------------------------------------------------------------------------
extern "C" void kernel_launch(void* const* d_in, const int* in_sizes, int n_in,
                              void* d_out, int out_size, void* d_ws, size_t ws_size,
                              hipStream_t stream) {
    const float* student = (const float*)d_in[0];
    const float* teacher = (const float*)d_in[1];
    const int*   idx1    = (const int*)d_in[2];
    const int*   idx2    = (const int*)d_in[3];
    const int NG = in_sizes[2];
    float* out = (float*)d_out;

    // workspace layout: Dt(64MB) | Ds(64MB) | tn bf16(6MB) | sA(1MB) | acc(8 floats)
    float* Dt = (float*)d_ws;
    float* Ds = Dt + (size_t)B_N * B_N;
    unsigned short* tn = (unsigned short*)(Ds + (size_t)B_N * B_N);
    float* sA = (float*)(tn + (size_t)B_N * D_T);
    float* acc = sA + (size_t)B_N * D_S;

    zero_acc_kernel<<<1, 32, 0, stream>>>(acc);
    normalize_teacher_kernel<<<B_N, 256, 0, stream>>>(teacher, tn);
    prep_student_a_kernel<<<(B_N * D_S + 255) / 256, 256, 0, stream>>>(student, sA);

    // 128x128 wave-tiles (32x32 output per wave), 4 waves per block
    teacher_gram_kernel<<<(128 * 128) / 4, 128, 0, stream>>>(tn, Dt);
    student_gram_kernel<<<(128 * 128) / 4, 128, 0, stream>>>(sA, student, Ds);

    topk_kl_kernel<<<B_N, 32, 0, stream>>>(Dt, Ds, acc);
    global_pairs_kernel<<<1024, 256, 0, stream>>>(idx1, idx2, Dt, Ds, acc, NG);
    finalize_kernel<<<1, 1, 0, stream>>>(acc, NG, out);
}
